// Messages_6167573037369
// MI455X (gfx1250) — matrix operand order repaired and checked by reading, not compiled, not run on hardware
//
#include <hip/hip_runtime.h>

// ---------------------------------------------------------------------------
// Equivariant message passing on MI455X (gfx1250), fused bf16-WMMA version.
// Branch-free per-term H-tile build; component-major (transposed) xv/xd LDS
// copies so every packed-pair operand comes from one aligned ds_load_b64.
// Scheduler fenced at term boundaries to bound VGPR pressure (<256).
// Three GEMMs (psi_a 16x448x32, psi_v 48x704x16, psi_d 144x640x8) per
// 16-edge tile; P pre-swizzled into bf16 WMMA B fragments in d_ws.
// ---------------------------------------------------------------------------

typedef __attribute__((ext_vector_type(16))) __bf16 v16bf;
typedef __attribute__((ext_vector_type(2)))  __bf16 bf16x2;
typedef __attribute__((ext_vector_type(8)))  float  v8f;

#define KTA 14
#define KTV 22
#define KTD 20

// d_ws dword offsets of B fragments (each tile = 256 dwords)
#define WSV (KTA * 2 * 256)   // 7168
#define WSD (WSV + KTV * 256) // 12800
#define WSTOT (WSD + KTD * 256)

// per-edge LDS float offsets
#define PHIA 0
#define RHAT 8
#define PHIV 12
#define PHID 36
#define XAo  108   // x_a[b],          32 floats, row-major (pairs adjacent)
#define XVT  140   // xvT[c][b] (3x16), component-major transpose of x_v
#define XDT  188   // xdT[r][b] (9x8),  component-major transpose of x_d
#define ESTR 260
#define WAVES 2

__device__ __forceinline__ void wbar() { __builtin_amdgcn_wave_barrier(); }

#if __has_builtin(__builtin_amdgcn_sched_barrier)
#define SBAR() __builtin_amdgcn_sched_barrier(0)
#else
#define SBAR() __builtin_amdgcn_wave_barrier()
#endif

__device__ __forceinline__ unsigned pk2(float a, float b) {
  union { __bf16 h[2]; unsigned u; } p;
  p.h[0] = (__bf16)a; p.h[1] = (__bf16)b;
  return p.u;
}
__device__ __forceinline__ unsigned mul2(unsigned a, unsigned b) { // v_pk_mul_bf16
  union { bf16x2 v; unsigned u; } x, y, r;
  x.u = a; y.u = b;
  r.v = x.v * y.v;
  return r.u;
}
__device__ __forceinline__ float2 f2ld(const float* p) { return *(const float2*)p; }
__device__ __forceinline__ unsigned pk2f2(const float* p) { float2 t = f2ld(p); return pk2(t.x, t.y); }
__device__ __forceinline__ float2 f2mul(float s, float2 a) { return make_float2(s*a.x, s*a.y); }
__device__ __forceinline__ float2 f2fma(float s, float2 a, float2 c) {
  return make_float2(fmaf(s, a.x, c.x), fmaf(s, a.y, c.y));
}

// -------- prep: swizzle P tensors into bf16 WMMA B fragments in d_ws -------
__global__ __launch_bounds__(256) void prep_kernel(
    const float* __restrict__ P000, const float* __restrict__ P110, const float* __restrict__ P220,
    const float* __restrict__ P011, const float* __restrict__ P101, const float* __restrict__ P121,
    const float* __restrict__ P211, const float* __restrict__ P111, const float* __restrict__ P022,
    const float* __restrict__ P202, const float* __restrict__ P112, const float* __restrict__ P222,
    const float* __restrict__ P212, unsigned* __restrict__ ws)
{
  int idx = blockIdx.x * 256 + threadIdx.x;
  if (idx >= WSTOT) return;
  int gemm, ktile, ntile, lrem;
  if (idx < WSV)      { gemm = 0; int dw = idx;        ktile = dw >> 9; int r = dw & 511; ntile = r >> 8; lrem = r & 255; }
  else if (idx < WSD) { gemm = 1; int dw = idx - WSV;  ktile = dw >> 8; ntile = 0; lrem = dw & 255; }
  else                { gemm = 2; int dw = idx - WSD;  ktile = dw >> 8; ntile = 0; lrem = dw & 255; }
  int lane = lrem >> 3, v = lrem & 7;
  int col = ntile * 16 + (lane & 15);
  int k0 = ktile * 32 + ((lane < 16) ? 0 : 16) + 2 * v;  // B layout: lanes0-15 K0-15, lanes16-31 K16-31
  float bv[2] = {0.f, 0.f};
  #pragma unroll
  for (int t = 0; t < 2; ++t) {
    int k = k0 + t; float val = 0.f;
    if (gemm == 0) {
      if (k < 256)      { int l = k >> 5, b = k & 31;              val = P000[(col*8 + l)*32 + b]; }
      else if (k < 384) { int u = k - 256, l = u >> 4, b = u & 15; val = P110[(col*8 + l)*16 + b]; }
      else              { int u = k - 384, l = u >> 3, b = u & 7;  val = P220[(col*8 + l)*8  + b]; }
    } else if (gemm == 1) {
      if (k < 128)      { int l = k >> 4, b = k & 15;              val = P011[(col*8 + l)*16 + b]; }
      else if (k < 384) { int u = k - 128, l = u >> 5, b = u & 31; val = P101[(col*8 + l)*32 + b]; }
      else if (k < 448) { int u = k - 384, l = u >> 3, b = u & 7;  val = P121[(col*8 + l)*8  + b]; }
      else if (k < 576) { int u = k - 448, l = u >> 4, b = u & 15; val = P211[(col*8 + l)*16 + b]; }
      else              { int u = k - 576, l = u >> 4, b = u & 15; val = P111[(col*8 + l)*16 + b]; }
    } else if (col < 8) {
      if (k < 64)       { int l = k >> 3, b = k & 7;               val = P022[(col*8 + l)*8  + b]; }
      else if (k < 320) { int u = k - 64,  l = u >> 5, b = u & 31; val = P202[(col*8 + l)*32 + b]; }
      else if (k < 448) { int u = k - 320, l = u >> 4, b = u & 15; val = P112[(col*8 + l)*16 + b]; }
      else if (k < 512) { int u = k - 448, l = u >> 3, b = u & 7;  val = P222[(col*8 + l)*8  + b]; }
      else              { int u = k - 512, l = u >> 4, b = u & 15; val = P212[(col*8 + l)*16 + b]; }
    }
    bv[t] = val;
  }
  ws[idx] = pk2(bv[0], bv[1]);
}

// ----------------------------- edge kernel ---------------------------------
__global__ __launch_bounds__(64) void edge_kernel(
    const float* __restrict__ r_ij, const float* __restrict__ x_a,
    const float* __restrict__ x_v,  const float* __restrict__ x_d,
    const int* __restrict__ src, const int* __restrict__ dst,
    const uint4* __restrict__ bfrag,
    float* __restrict__ outA, float* __restrict__ outV, float* __restrict__ outD,
    int nEdges)
{
  __shared__ float    sEdge[WAVES][16][ESTR];
  __shared__ unsigned sTile[WAVES][2][16][20];   // ping-pong A-tile buffers

  const int wid  = threadIdx.x >> 5;
  const int lane = threadIdx.x & 31;
  const int base = (blockIdx.x * WAVES + wid) * 16;
  if (base >= nEdges) return;
  float (*ED)[ESTR] = sEdge[wid];
  const int m  = lane & 15;
  const int hi = lane >> 4;

  // ---- stage 1: radial encoding + squashed direction (lanes 0..15) ----
  if (lane < 16) {
    int e = base + lane; if (e >= nEdges) e = nEdges - 1;
    float rx = r_ij[e*3], ry = r_ij[e*3+1], rz = r_ij[e*3+2];
    float x2  = (rx*rx + ry*ry + rz*rz) * (1.0f / 2.5f);
    float s   = sqrtf(x2);
    float env = fmaxf(0.f, 1.f - x2);
    #pragma unroll
    for (int l = 0; l < 8; ++l)
      ED[lane][PHIA + l] = cosf(3.14159265358979323846f * (float)l * s) * env;
    float c = 7.0f / 2.5f;
    float yx = rx*c, yy = ry*c, yz = rz*c;
    float inv = rsqrtf(1.f + yx*yx + yy*yy + yz*yz);
    ED[lane][RHAT+0] = yx*inv; ED[lane][RHAT+1] = yy*inv; ED[lane][RHAT+2] = yz*inv;
  }
  wbar();

  // ---- stage 2: expand phi_v (8x3), phi_d (8x3x3); slot=m, half=hi ----
  {
    const float* e = ED[m];
    float rh0 = e[RHAT], rh1 = e[RHAT+1], rh2 = e[RHAT+2];
    #pragma unroll
    for (int l = 0; l < 4; ++l) {
      int lg = hi * 4 + l;
      float en = e[PHIA + lg];
      float pv0 = en*rh0, pv1 = en*rh1, pv2 = en*rh2;
      ED[m][PHIV + 3*lg + 0] = pv0;
      ED[m][PHIV + 3*lg + 1] = pv1;
      ED[m][PHIV + 3*lg + 2] = pv2;
      ED[m][PHID + 9*lg + 0] = pv0*rh0; ED[m][PHID + 9*lg + 1] = pv0*rh1; ED[m][PHID + 9*lg + 2] = pv0*rh2;
      ED[m][PHID + 9*lg + 3] = pv1*rh0; ED[m][PHID + 9*lg + 4] = pv1*rh1; ED[m][PHID + 9*lg + 5] = pv1*rh2;
      ED[m][PHID + 9*lg + 6] = pv2*rh0; ED[m][PHID + 9*lg + 7] = pv2*rh1; ED[m][PHID + 9*lg + 8] = pv2*rh2;
    }
  }

  // ---- stage 3: gather dst-node features; transpose xv/xd on the fly ----
  {
    int eidx = base + m; if (eidx >= nEdges) eidx = nEdges - 1;
    int node = dst[eidx];
    const float4* pa4 = (const float4*)(x_a + node*32);
    const float4* pv4 = (const float4*)(x_v + node*48);
    const float4* pd4 = (const float4*)(x_d + node*72);
    #pragma unroll
    for (int c = 0; c < 4; ++c) *(float4*)&ED[m][XAo + (hi*4 + c)*4] = pa4[hi*4 + c];
    // lane's 6 float4 chunks of x_v cover rows b = 8*hi .. 8*hi+7 (all comps)
    float xvloc[24];
    #pragma unroll
    for (int c = 0; c < 6; ++c) {
      float4 t = pv4[hi*6 + c];
      xvloc[4*c+0]=t.x; xvloc[4*c+1]=t.y; xvloc[4*c+2]=t.z; xvloc[4*c+3]=t.w;
    }
    #pragma unroll
    for (int c = 0; c < 3; ++c)
      #pragma unroll
      for (int bb = 0; bb < 8; ++bb)
        ED[m][XVT + c*16 + hi*8 + bb] = xvloc[3*bb + c];
    // lane's 9 float4 chunks of x_d cover rows b = 4*hi .. 4*hi+3 (all comps)
    float xdloc[36];
    #pragma unroll
    for (int c = 0; c < 9; ++c) {
      float4 t = pd4[hi*9 + c];
      xdloc[4*c+0]=t.x; xdloc[4*c+1]=t.y; xdloc[4*c+2]=t.z; xdloc[4*c+3]=t.w;
    }
    #pragma unroll
    for (int r = 0; r < 9; ++r)
      #pragma unroll
      for (int bb = 0; bb < 4; ++bb)
        ED[m][XDT + r*8 + hi*4 + bb] = xdloc[9*bb + r];
  }
  wbar();

  union AF { uint4 q[2]; v16bf v; };
  auto loadAfrag = [&](int par) -> v16bf {
    AF af;
    af.q[0] = *(const uint4*)&sTile[wid][par][m][hi*8];      // K 0-7  / 16-23
    af.q[1] = *(const uint4*)&sTile[wid][par][m][hi*8 + 4];  // K 8-15 / 24-31
    return af.v;
  };
  auto loadBfrag = [&](int tile) -> v16bf {
    AF bf;
    const uint4* p = bfrag + (size_t)(tile*32 + lane)*2;
    bf.q[0] = p[0]; bf.q[1] = p[1];
    return bf.v;
  };
  auto flushTile = [&](int par, const unsigned* w) {
    uint4 lo = {w[0], w[1], w[2], w[3]};
    uint4 hh = {w[4], w[5], w[6], w[7]};
    *(uint4*)&sTile[wid][par][m][hi*8]     = lo;
    *(uint4*)&sTile[wid][par][m][hi*8 + 4] = hh;
    wbar();
  };
  auto mma = [&](v8f& acc, v16bf a, int tile) {
    acc = __builtin_amdgcn_wmma_f32_16x16x32_bf16(false, a, false, loadBfrag(tile), (short)0, acc, false, false);
  };

  // =================== GEMM 1: psi_a (16 x 448 x 32) ===================
  {
    v8f acc0 = {0,0,0,0,0,0,0,0}, acc1 = {0,0,0,0,0,0,0,0};
    const float* e = ED[m];
    unsigned w[8];
    { // T000: tiles 0..7  H = phi_a[l]*xa[b] (bf16 path)
      unsigned xab[8];
      #pragma unroll
      for (int t = 0; t < 8; ++t) xab[t] = pk2f2(&e[XAo + hi*16 + 2*t]);
      #pragma unroll
      for (int ktl = 0; ktl < 8; ++ktl) {
        unsigned pp = pk2(e[PHIA + ktl], e[PHIA + ktl]);
        #pragma unroll
        for (int v = 0; v < 8; ++v) w[v] = mul2(pp, xab[v]);
        flushTile(ktl & 1, w);
        v16bf a = loadAfrag(ktl & 1);
        mma(acc0, a, 2*ktl); mma(acc1, a, 2*ktl + 1);
      }
    }
    SBAR();
    { // T110: tiles 8..11  H = dot3(phi_v[l], xv[b]), l=2ktl+hi, b=2v
      float2 xvp[3][8];
      #pragma unroll
      for (int c = 0; c < 3; ++c)
        #pragma unroll
        for (int v = 0; v < 8; ++v) xvp[c][v] = f2ld(&e[XVT + c*16 + 2*v]);
      #pragma unroll
      for (int ktl = 0; ktl < 4; ++ktl) {
        int l = 2*ktl + hi;
        float p0 = e[PHIV+3*l], p1 = e[PHIV+3*l+1], p2 = e[PHIV+3*l+2];
        #pragma unroll
        for (int v = 0; v < 8; ++v) {
          float2 h = f2mul(p0, xvp[0][v]);
          h = f2fma(p1, xvp[1][v], h);
          h = f2fma(p2, xvp[2][v], h);
          w[v] = pk2(h.x, h.y);
        }
        flushTile(ktl & 1, w);
        v16bf a = loadAfrag(ktl & 1);
        mma(acc0, a, 2*(8+ktl)); mma(acc1, a, 2*(8+ktl) + 1);
      }
    }
    SBAR();
    { // T220: tiles 12..13  H = dot9(phi_d[l], xd[b]), l=4ktl+2hi+(v>>2), b=2(v&3)
      float2 xdp[9][4];
      #pragma unroll
      for (int r = 0; r < 9; ++r)
        #pragma unroll
        for (int c = 0; c < 4; ++c) xdp[r][c] = f2ld(&e[XDT + r*8 + 2*c]);
      #pragma unroll
      for (int ktl = 0; ktl < 2; ++ktl) {
        int l0 = 4*ktl + 2*hi;
        float pd0[9], pd1[9];
        #pragma unroll
        for (int r = 0; r < 9; ++r) { pd0[r] = e[PHID + 9*l0 + r]; pd1[r] = e[PHID + 9*(l0+1) + r]; }
        #pragma unroll
        for (int v = 0; v < 8; ++v) {
          int c = v & 3;
          float2 h = f2mul((v < 4) ? pd0[0] : pd1[0], xdp[0][c]);
          #pragma unroll
          for (int r = 1; r < 9; ++r) h = f2fma((v < 4) ? pd0[r] : pd1[r], xdp[r][c], h);
          w[v] = pk2(h.x, h.y);
        }
        flushTile(ktl & 1, w);
        v16bf a = loadAfrag(ktl & 1);
        mma(acc0, a, 2*(12+ktl)); mma(acc1, a, 2*(12+ktl) + 1);
      }
    }
    SBAR();
    #pragma unroll
    for (int g = 0; g < 8; ++g) {
      int eidx = base + g + hi*8;
      if (eidx < nEdges) {
        int node = src[eidx];
        unsafeAtomicAdd(&outA[node*32 + m],      0.1f*acc0[g]);
        unsafeAtomicAdd(&outA[node*32 + 16 + m], 0.1f*acc1[g]);
      }
    }
  }

  // =================== GEMM 2: psi_v (48 x 704 x 16) ===================
  for (int rt = 0; rt < 3; ++rt) {
    v8f acc = {0,0,0,0,0,0,0,0};
    int rowm = rt*16 + m, slot = rowm/3, bi = rowm - 3*slot;
    int i1 = (bi == 2) ? 0 : bi + 1, i2 = (bi == 0) ? 2 : bi - 1;
    const float* e = ED[slot];
    unsigned w[8];
    { // T011: tiles 0..3  H = phi_a[l]*xv[b,i] (bf16 path)
      unsigned xvib[8];
      #pragma unroll
      for (int t = 0; t < 8; ++t) xvib[t] = pk2f2(&e[XVT + bi*16 + 2*t]);
      #pragma unroll
      for (int ktl = 0; ktl < 4; ++ktl) {
        float pa = e[PHIA + 2*ktl + hi];
        unsigned pp = pk2(pa, pa);
        #pragma unroll
        for (int v = 0; v < 8; ++v) w[v] = mul2(pp, xvib[v]);
        flushTile(ktl & 1, w);
        mma(acc, loadAfrag(ktl & 1), WSV/256 + ktl);
      }
    }
    SBAR();
    { // T101: tiles 4..11  H = phi_v[l,i]*xa[b] (bf16 path)
      unsigned xab[8];
      #pragma unroll
      for (int t = 0; t < 8; ++t) xab[t] = pk2f2(&e[XAo + hi*16 + 2*t]);
      #pragma unroll
      for (int ktl = 0; ktl < 8; ++ktl) {
        float pv = e[PHIV + 3*ktl + bi];
        unsigned pp = pk2(pv, pv);
        #pragma unroll
        for (int v = 0; v < 8; ++v) w[v] = mul2(pp, xab[v]);
        flushTile(ktl & 1, w);
        mma(acc, loadAfrag(ktl & 1), WSV/256 + 4 + ktl);
      }
    }
    SBAR();
    { // T121: tiles 12..13  H = sum_j phi_v[l,j]*xd[b,j,i], b=2(v&3)
      float2 xdpi[3][4];
      #pragma unroll
      for (int jj = 0; jj < 3; ++jj)
        #pragma unroll
        for (int c = 0; c < 4; ++c) xdpi[jj][c] = f2ld(&e[XDT + (3*jj + bi)*8 + 2*c]);
      #pragma unroll
      for (int ktl = 0; ktl < 2; ++ktl) {
        int l0 = 4*ktl + 2*hi;
        float pv0[3], pv1[3];
        #pragma unroll
        for (int r = 0; r < 3; ++r) { pv0[r] = e[PHIV + 3*l0 + r]; pv1[r] = e[PHIV + 3*(l0+1) + r]; }
        #pragma unroll
        for (int v = 0; v < 8; ++v) {
          int c = v & 3;
          float2 h = f2mul((v<4)?pv0[0]:pv1[0], xdpi[0][c]);
          h = f2fma((v<4)?pv0[1]:pv1[1], xdpi[1][c], h);
          h = f2fma((v<4)?pv0[2]:pv1[2], xdpi[2][c], h);
          w[v] = pk2(h.x, h.y);
        }
        flushTile(ktl & 1, w);
        mma(acc, loadAfrag(ktl & 1), WSV/256 + 12 + ktl);
      }
    }
    SBAR();
    { // T211: tiles 14..17  H = dot3(phi_d[l,i,:], xv[b]), l=2ktl+hi, b=2v
      float2 xvp[3][8];
      #pragma unroll
      for (int c = 0; c < 3; ++c)
        #pragma unroll
        for (int v = 0; v < 8; ++v) xvp[c][v] = f2ld(&e[XVT + c*16 + 2*v]);
      #pragma unroll
      for (int ktl = 0; ktl < 4; ++ktl) {
        int l = 2*ktl + hi;
        float p0 = e[PHID+9*l+3*bi], p1 = e[PHID+9*l+3*bi+1], p2 = e[PHID+9*l+3*bi+2];
        #pragma unroll
        for (int v = 0; v < 8; ++v) {
          float2 h = f2mul(p0, xvp[0][v]);
          h = f2fma(p1, xvp[1][v], h);
          h = f2fma(p2, xvp[2][v], h);
          w[v] = pk2(h.x, h.y);
        }
        flushTile(ktl & 1, w);
        mma(acc, loadAfrag(ktl & 1), WSV/256 + 14 + ktl);
      }
    }
    SBAR();
    { // T111: tiles 18..21  H = (phi_v[l] x xv[b])_i, l=2ktl+hi, b=2v
      float2 x1p[8], x2p[8];
      #pragma unroll
      for (int v = 0; v < 8; ++v) { x1p[v] = f2ld(&e[XVT + i1*16 + 2*v]); x2p[v] = f2ld(&e[XVT + i2*16 + 2*v]); }
      #pragma unroll
      for (int ktl = 0; ktl < 4; ++ktl) {
        int l = 2*ktl + hi;
        float a1 = e[PHIV + 3*l + i1], a2 = e[PHIV + 3*l + i2];
        #pragma unroll
        for (int v = 0; v < 8; ++v) {
          float2 h = f2mul(a1, x2p[v]);
          h = f2fma(-a2, x1p[v], h);
          w[v] = pk2(h.x, h.y);
        }
        flushTile(ktl & 1, w);
        mma(acc, loadAfrag(ktl & 1), WSV/256 + 18 + ktl);
      }
    }
    SBAR();
    #pragma unroll
    for (int g = 0; g < 8; ++g) {
      int row = rt*16 + g + hi*8;
      int sl = row/3, ii = row - 3*sl, eidx = base + sl;
      if (eidx < nEdges) unsafeAtomicAdd(&outV[src[eidx]*48 + m*3 + ii], 0.1f*acc[g]);
    }
  }

  // =================== GEMM 3: psi_d (144 x 640 x 8) ===================
  for (int rt = 0; rt < 9; ++rt) {
    v8f acc = {0,0,0,0,0,0,0,0};
    int rowm = rt*16 + m, slot = rowm/9, rem = rowm - 9*slot, bi = rem/3, bj = rem - 3*bi;
    int j1 = (bj == 2) ? 0 : bj + 1, j2 = (bj == 0) ? 2 : bj - 1;
    const float* e = ED[slot];
    unsigned w[8];
    { // T022: tiles 0..1  H = phi_a[l]*xd[b,i,j], b=2(v&3) (bf16 path)
      unsigned xdijb[4];
      #pragma unroll
      for (int c = 0; c < 4; ++c) xdijb[c] = pk2f2(&e[XDT + (3*bi + bj)*8 + 2*c]);
      #pragma unroll
      for (int ktl = 0; ktl < 2; ++ktl) {
        int l0 = 4*ktl + 2*hi;
        unsigned pp0 = pk2(e[PHIA + l0], e[PHIA + l0]);
        unsigned pp1 = pk2(e[PHIA + l0 + 1], e[PHIA + l0 + 1]);
        #pragma unroll
        for (int v = 0; v < 8; ++v) w[v] = mul2((v < 4) ? pp0 : pp1, xdijb[v & 3]);
        flushTile(ktl & 1, w);
        mma(acc, loadAfrag(ktl & 1), WSD/256 + ktl);
      }
    }
    SBAR();
    { // T202: tiles 2..9  H = phi_d[l,i,j]*xa[b] (bf16 path)
      unsigned xab[8];
      #pragma unroll
      for (int t = 0; t < 8; ++t) xab[t] = pk2f2(&e[XAo + hi*16 + 2*t]);
      #pragma unroll
      for (int ktl = 0; ktl < 8; ++ktl) {
        float pd = e[PHID + 9*ktl + 3*bi + bj];
        unsigned pp = pk2(pd, pd);
        #pragma unroll
        for (int v = 0; v < 8; ++v) w[v] = mul2(pp, xab[v]);
        flushTile(ktl & 1, w);
        mma(acc, loadAfrag(ktl & 1), WSD/256 + 2 + ktl);
      }
    }
    SBAR();
    { // T112: tiles 10..13  H = phi_v[l,i]*xv[b,j] (bf16 path)
      unsigned xvjb[8];
      #pragma unroll
      for (int t = 0; t < 8; ++t) xvjb[t] = pk2f2(&e[XVT + bj*16 + 2*t]);
      #pragma unroll
      for (int ktl = 0; ktl < 4; ++ktl) {
        float pv = e[PHIV + 3*(2*ktl + hi) + bi];
        unsigned pp = pk2(pv, pv);
        #pragma unroll
        for (int v = 0; v < 8; ++v) w[v] = mul2(pp, xvjb[v]);
        flushTile(ktl & 1, w);
        mma(acc, loadAfrag(ktl & 1), WSD/256 + 10 + ktl);
      }
    }
    SBAR();
    { // T222: tiles 14..15  H = sum_p phi_d[l,i,p]*xd[b,p,j], b=2(v&3)
      float2 xdpj[3][4];
      #pragma unroll
      for (int p = 0; p < 3; ++p)
        #pragma unroll
        for (int c = 0; c < 4; ++c) xdpj[p][c] = f2ld(&e[XDT + (3*p + bj)*8 + 2*c]);
      #pragma unroll
      for (int ktl = 0; ktl < 2; ++ktl) {
        int l0 = 4*ktl + 2*hi;
        float pd0[3], pd1[3];
        #pragma unroll
        for (int r = 0; r < 3; ++r) { pd0[r] = e[PHID + 9*l0 + 3*bi + r]; pd1[r] = e[PHID + 9*(l0+1) + 3*bi + r]; }
        #pragma unroll
        for (int v = 0; v < 8; ++v) {
          int c = v & 3;
          float2 h = f2mul((v<4)?pd0[0]:pd1[0], xdpj[0][c]);
          h = f2fma((v<4)?pd0[1]:pd1[1], xdpj[1][c], h);
          h = f2fma((v<4)?pd0[2]:pd1[2], xdpj[2][c], h);
          w[v] = pk2(h.x, h.y);
        }
        flushTile(ktl & 1, w);
        mma(acc, loadAfrag(ktl & 1), WSD/256 + 14 + ktl);
      }
    }
    SBAR();
    { // T212: tiles 16..19  H = (phi_d[l,i,:] x xv[b])_j, l=2ktl+hi, b=2v
      float2 x1p[8], x2p[8];
      #pragma unroll
      for (int v = 0; v < 8; ++v) { x1p[v] = f2ld(&e[XVT + j1*16 + 2*v]); x2p[v] = f2ld(&e[XVT + j2*16 + 2*v]); }
      #pragma unroll
      for (int ktl = 0; ktl < 4; ++ktl) {
        int l = 2*ktl + hi;
        float a1 = e[PHID + 9*l + 3*bi + j1], a2 = e[PHID + 9*l + 3*bi + j2];
        #pragma unroll
        for (int v = 0; v < 8; ++v) {
          float2 h = f2mul(a1, x2p[v]);
          h = f2fma(-a2, x1p[v], h);
          w[v] = pk2(h.x, h.y);
        }
        flushTile(ktl & 1, w);
        mma(acc, loadAfrag(ktl & 1), WSD/256 + 16 + ktl);
      }
    }
    SBAR();
    if (m < 8) {
      #pragma unroll
      for (int g = 0; g < 8; ++g) {
        int row = rt*16 + g + hi*8;
        int sl = row/9, r9 = row - 9*sl, ii = r9/3, jj = r9 - 3*ii, eidx = base + sl;
        if (eidx < nEdges) unsafeAtomicAdd(&outD[src[eidx]*72 + m*9 + ii*3 + jj], 0.1f*acc[g]);
      }
    }
  }
}

// ---------------------------------------------------------------------------
extern "C" void kernel_launch(void* const* d_in, const int* in_sizes, int n_in,
                              void* d_out, int out_size, void* d_ws, size_t ws_size,
                              hipStream_t stream) {
  const float* r_ij = (const float*)d_in[0];
  const float* x_a  = (const float*)d_in[1];
  const float* x_v  = (const float*)d_in[2];
  const float* x_d  = (const float*)d_in[3];
  const float* P[13];
  for (int i = 0; i < 13; ++i) P[i] = (const float*)d_in[4 + i];
  const int* src = (const int*)d_in[17];
  const int* dst = (const int*)d_in[18];

  int E = in_sizes[0] / 3;
  int N = in_sizes[1] / 32;
  float* out  = (float*)d_out;
  float* outA = out;
  float* outV = out + (size_t)N * 32;
  float* outD = outV + (size_t)N * 48;

  hipMemsetAsync(d_out, 0, (size_t)out_size * sizeof(float), stream);

  unsigned* ws = (unsigned*)d_ws;
  prep_kernel<<<(WSTOT + 255) / 256, 256, 0, stream>>>(
      P[0], P[1], P[2], P[3], P[4], P[5], P[6], P[7], P[8], P[9], P[10], P[11], P[12], ws);

  int nblk = (E + 16 * WAVES - 1) / (16 * WAVES);
  edge_kernel<<<nblk, 64, 0, stream>>>(r_ij, x_a, x_v, x_d, src, dst,
                                       (const uint4*)ws, outA, outV, outD, E);
}
